// CoherenceAttention_2783138808278
// MI455X (gfx1250) — matrix operand order repaired and checked by reading, not compiled, not run on hardware
//
#include <hip/hip_runtime.h>
#include <hip/hip_bf16.h>
#include <cstdint>
#include <cstddef>

typedef unsigned short ushort_t;
typedef unsigned int uint_t;

typedef __attribute__((ext_vector_type(16))) __bf16 v16bf;
typedef __attribute__((ext_vector_type(8)))  float  v8f;
typedef __attribute__((ext_vector_type(4)))  unsigned int v4u;
typedef __attribute__((ext_vector_type(8)))  int v8i_t;
typedef __attribute__((ext_vector_type(4)))  int v4i_t;

#define D_MODELC 512
#define DKC      32
#define HIDC     512
#define BBC      4
#define TTC      512
// M1 = BBC*DKC*TTC = 65536 rows for every big GEMM

// TDM path: only when the gfx1250 builtins exist in the device pass.
// This toolchain (clang-23 / therock-10.0) uses the 6-arg form:
//   (uint32x4 g0, int32x8 g1, int32x4 g2, int32x4 g3, int32x8 g4, i32 cpol)
#if defined(__HIP_DEVICE_COMPILE__) && defined(__has_builtin)
#if __has_builtin(__builtin_amdgcn_tensor_load_to_lds) && \
    __has_builtin(__builtin_amdgcn_s_wait_tensorcnt)
#define USE_TDM 1
#endif
#endif
#ifndef USE_TDM
#define USE_TDM 0
#endif

__device__ __forceinline__ ushort_t f2bf(float f) {
    uint_t u = __float_as_uint(f);
    u += 0x7FFFu + ((u >> 16) & 1u);        // round-to-nearest-even
    return (ushort_t)(u >> 16);
}
__device__ __forceinline__ float bf2f(ushort_t h) {
    return __uint_as_float(((uint_t)h) << 16);
}

// ---------------------------------------------------------------------------
// fp32 -> bf16 convert (grid-stride)
// ---------------------------------------------------------------------------
__global__ void cvt_f32_bf16_kernel(const float* __restrict__ src,
                                    ushort_t* __restrict__ dst, int n) {
    for (int i = blockIdx.x * blockDim.x + threadIdx.x; i < n;
         i += gridDim.x * blockDim.x)
        dst[i] = f2bf(src[i]);
}

// ---------------------------------------------------------------------------
// trace(softmax(var_attn)) -> single float in ws
// ---------------------------------------------------------------------------
__global__ void zero_scalar_kernel(float* p) {
    if (threadIdx.x == 0 && blockIdx.x == 0) *p = 0.0f;
}

__global__ void __launch_bounds__(512)
trace_softmax_kernel(const float* __restrict__ A, float* __restrict__ tr) {
    const int i = threadIdx.x;                 // row, one block of 512
    float m = -1e30f;
    for (int j = 0; j < D_MODELC; ++j) m = fmaxf(m, A[(size_t)i * D_MODELC + j]);
    float s = 0.0f;
    for (int j = 0; j < D_MODELC; ++j) s += expf(A[(size_t)i * D_MODELC + j] - m);
    float diag = expf(A[(size_t)i * D_MODELC + i] - m) / s;
    atomicAdd(tr, diag);
}

// ---------------------------------------------------------------------------
// Generic bf16 WMMA GEMM: C = A[M,K] * B[K,N] + bias, fp32 accumulate.
// Block tile 64(M) x 128(N), K-step 64. 256 threads = 8 waves; wave (wm,wn)
// owns a 16x64 output strip -> 4 f32 accumulators, 8 WMMAs per K-step.
// A panel staged into LDS by the Tensor Data Mover (one tensor_load_to_lds
// per block per K-step, issued by wave 0, completed with s_wait_tensorcnt);
// B tile staged transposed [n][k] so fragments are contiguous b128 LDS reads.
// Epilogues: 0 = qkv split-store (N=1536, interleave-3 -> q/k/v bf16)
//            1 = exact GELU -> bf16
//            2 = + residual  -> bf16
//            3 = final fp32 store with (b,t,n,:) -> (b,n,t,:) transpose
// ---------------------------------------------------------------------------
enum { EPI_QKV = 0, EPI_GELU = 1, EPI_RESID = 2, EPI_OUT = 3 };

#define BM 64
#define BN 128
#define BK 64

template <int EPI>
__global__ void __launch_bounds__(256)
gemm_bf16_kernel(const ushort_t* __restrict__ A,
                 const ushort_t* __restrict__ Bm,
                 const float* __restrict__ bias,
                 int M, int N, int K,
                 ushort_t* __restrict__ outQ,
                 ushort_t* __restrict__ outK,
                 ushort_t* __restrict__ outV,
                 ushort_t* __restrict__ outBf,
                 const ushort_t* __restrict__ resid,
                 float* __restrict__ outF) {
    __shared__ __align__(16) ushort_t As[BM * BK];   // [row][k]   8 KB
    __shared__ __align__(16) ushort_t Bs[BN * BK];   // [n][k]    16 KB

    const int tid  = threadIdx.x;
    const int lane = tid & 31;
    const int w    = tid >> 5;     // 0..7
    const int wm   = w & 3;        // 16-row strip within 64
    const int wn   = w >> 2;       // 0..1 : 64-col half within 128

    const int bm0 = blockIdx.y * BM;
    const int bn0 = blockIdx.x * BN;

    v8f cacc[4] = {};

    const int lhalf = lane >> 4;          // 0/1
    const int l15   = lane & 15;
    const int mr    = wm * 16 + l15;
    const int cb    = wn * 64;

    for (int k0 = 0; k0 < K; k0 += BK) {
        __syncthreads();   // everyone done reading the previous tiles

#if USE_TDM
        if (w == 0) {
            // --- Tensor DMA descriptor (D#) for the A tile -----------------
            // 2D tile: tile_dim0 = BK elems along K (contiguous),
            //          tile_dim1 = BM rows, row stride = K elements.
            const unsigned long long ga =
                (unsigned long long)(uintptr_t)(A + (size_t)bm0 * K + k0);
            const unsigned int ldsA = (unsigned int)(uintptr_t)(&As[0]);
            v4u g0;
            g0.x = 1u;                                   // count=1, user mode
            g0.y = ldsA;                                 // lds_addr [63:32]
            g0.z = (unsigned int)(ga & 0xFFFFFFFFu);     // global_addr [95:64]
            g0.w = (unsigned int)((ga >> 32) & 0x1FFFFFFu) | (2u << 30); // type=2
            const unsigned int td0 = (unsigned int)K;    // tensor_dim0
            const unsigned int td1 = (unsigned int)M;    // tensor_dim1
            const unsigned int st0 = (unsigned int)K;    // tensor_dim0_stride
            v8i_t g1;
            g1[0] = (int)(1u << 16);                     // data_size=1 (2B)
            g1[1] = (int)((td0 & 0xFFFFu) << 16);        // tensor_dim0 lo
            g1[2] = (int)((td0 >> 16) | ((td1 & 0xFFFFu) << 16));
            g1[3] = (int)((td1 >> 16) | ((unsigned)BK << 16));   // tile_dim0
            g1[4] = (int)BM;                             // tile_dim1 (tile_dim2=0)
            g1[5] = (int)st0;                            // stride0 lo
            g1[6] = 0;                                   // stride0 hi | stride1 lo
            g1[7] = 0;                                   // stride1 hi
            v4i_t gz4 = {};                              // groups 2/3: unused (2D)
            v8i_t gz8 = {};                              // extra group (zero-filled)
            __builtin_amdgcn_tensor_load_to_lds(g0, g1, gz4, gz4, gz8, 0);
        }
#else
        // manual A staging: 64x64 bf16, 2 x b128 per thread
#pragma unroll
        for (int i = 0; i < 2; ++i) {
            const int lin = tid + i * 256;               // 0..511
            const int row = lin >> 3;
            const int seg = (lin & 7) << 3;
            *(uint4*)&As[row * BK + seg] =
                *(const uint4*)&A[(size_t)(bm0 + row) * K + k0 + seg];
        }
#endif
        // B staging (transpose into [n][k]): 64x128 bf16, 4 x b128 per thread
#pragma unroll
        for (int i = 0; i < 4; ++i) {
            const int lin  = tid + i * 256;              // 0..1023
            const int kk   = lin >> 4;                   // 0..63
            const int nseg = (lin & 15) << 3;            // 0..120
            union { uint4 q; ushort_t u[8]; } t;
            t.q = *(const uint4*)&Bm[(size_t)(k0 + kk) * N + bn0 + nseg];
#pragma unroll
            for (int j = 0; j < 8; ++j) Bs[(nseg + j) * BK + kk] = t.u[j];
        }
        if (k0 + BK < K) {   // hint next B tile toward L2
            __builtin_prefetch(&Bm[(size_t)(k0 + BK + (tid >> 4)) * N + bn0], 0, 1);
        }
#if USE_TDM
        __builtin_amdgcn_s_wait_tensorcnt(0);  // wave0: TDM data landed in LDS
#endif
        __syncthreads();

        // two 16x16x32 sub-steps over the 64-wide K tile
#pragma unroll
        for (int ksub = 0; ksub < BK; ksub += 32) {
            union FR { v16bf v; uint4 q[2]; };
            FR fa;
            const int khA = ksub + (lhalf << 3);   // A: K halves {0..7}/{8..15}
            fa.q[0] = *(const uint4*)&As[mr * BK + khA];
            fa.q[1] = *(const uint4*)&As[mr * BK + khA + 16];
            const int khB = ksub + (lhalf << 3);   // B: K halves {0..7}/{8..15}
#pragma unroll
            for (int tc = 0; tc < 4; ++tc) {
                FR fb;
                const int nc = cb + tc * 16 + l15;
                fb.q[0] = *(const uint4*)&Bs[nc * BK + khB];
                fb.q[1] = *(const uint4*)&Bs[nc * BK + khB + 16];
                cacc[tc] = __builtin_amdgcn_wmma_f32_16x16x32_bf16(
                    false, fa.v, false, fb.v, (short)0, cacc[tc], false, false);
            }
        }
    }

    // epilogue: VGPR r -> M = r (+8 for lanes 16..31), N = lane&15
    const int mbase = bm0 + wm * 16 + (lhalf << 3);
#pragma unroll
    for (int tc = 0; tc < 4; ++tc) {
        const int gcol = bn0 + cb + tc * 16 + l15;
        const float bv = bias[gcol];
#pragma unroll
        for (int r = 0; r < 8; ++r) {
            const int grow = mbase + r;
            float val = cacc[tc][r] + bv;
            if (EPI == EPI_QKV) {
                const int which = gcol % 3;          // qkv interleave: h*3+c
                const int h     = gcol / 3;
                const size_t o  = (size_t)grow * HIDC + h;
                const ushort_t b16 = f2bf(val);
                if (which == 0)      outQ[o] = b16;
                else if (which == 1) outK[o] = b16;
                else                 outV[o] = b16;
            } else if (EPI == EPI_GELU) {
                const float g = 0.5f * val *
                                (1.0f + erff(val * 0.70710678118654752f));
                outBf[(size_t)grow * N + gcol] = f2bf(g);
            } else if (EPI == EPI_RESID) {
                const float res = bf2f(resid[(size_t)grow * N + gcol]);
                outBf[(size_t)grow * N + gcol] = f2bf(val + res);
            } else {  // EPI_OUT: rows are (b,t,n); output is [b, n, t, d]
                const int nidx = grow & 31;
                const int bt   = grow >> 5;
                const int t    = bt & 511;
                const int b    = bt >> 9;
                outF[((((size_t)b * DKC + nidx) * TTC) + t) * D_MODELC + gcol] = val;
            }
        }
    }
}

// ---------------------------------------------------------------------------
// Coherence + softmax over t + V scaling.
// rfft identities (exact for Pxx, Pyy, Re(Pxy); Im(Pxy) dropped):
//   sum_{k<=H/2} Xf conj(Yf) has Re = (H*dot(x,y) + X0*Y0 + Xn*Yn)/2
//   X0 = sum x[h], Xn = sum (-1)^h x[h]
// One block per (b,n); 512 threads, thread t owns signal pair (q,k)[:,t].
// Then out_var[(b,t,n),h] = trace_c * attn[t] * v[(b,n,t),h]  (bf16).
// ---------------------------------------------------------------------------
__global__ void __launch_bounds__(512)
coherence_kernel(const ushort_t* __restrict__ q,
                 const ushort_t* __restrict__ k,
                 const ushort_t* __restrict__ v,
                 const float* __restrict__ trp,
                 ushort_t* __restrict__ A2) {
    __shared__ float s_red[512];
    __shared__ float s_attn[512];

    const int bn = blockIdx.x;          // b*32 + n
    const int b  = bn >> 5;
    const int n  = bn & 31;
    const int t  = threadIdx.x;
    const size_t rbase = ((size_t)bn * TTC + t) * HIDC;

    float sqk = 0.f, sqq = 0.f, skk = 0.f;
    float q0 = 0.f, k0s = 0.f, qN = 0.f, kN = 0.f;
    for (int h8 = 0; h8 < HIDC / 8; ++h8) {
        union { uint4 q4; ushort_t u[8]; } uq, uk;
        uq.q4 = *(const uint4*)&q[rbase + h8 * 8];
        uk.q4 = *(const uint4*)&k[rbase + h8 * 8];
#pragma unroll
        for (int j = 0; j < 8; ++j) {
            const float a = bf2f(uq.u[j]);
            const float c = bf2f(uk.u[j]);
            sqk += a * c; sqq += a * a; skk += c * c;
            q0 += a; k0s += c;
            const float sgn = (j & 1) ? -1.f : 1.f;   // h parity == j parity
            qN += sgn * a; kN += sgn * c;
        }
    }
    const float inv257 = 1.0f / 257.0f;
    const float Pxy = 0.5f * (512.f * sqk + q0 * k0s + qN * kN) * inv257;
    const float Pxx = 0.5f * (512.f * sqq + q0 * q0 + qN * qN) * inv257;
    const float Pyy = 0.5f * (512.f * skk + k0s * k0s + kN * kN) * inv257;
    const float coh = (Pxy * Pxy) / fmaxf(Pxx * Pyy, 1e-6f);
    const float logit = coh * 5.6568542494923806f;   // / DK^-0.5

    // block softmax over t
    s_red[t] = logit; __syncthreads();
    for (int s = 256; s > 0; s >>= 1) {
        if (t < s) s_red[t] = fmaxf(s_red[t], s_red[t + s]);
        __syncthreads();
    }
    const float mx = s_red[0]; __syncthreads();
    const float e = __expf(logit - mx);
    s_red[t] = e; __syncthreads();
    for (int s = 256; s > 0; s >>= 1) {
        if (t < s) s_red[t] += s_red[t + s];
        __syncthreads();
    }
    s_attn[t] = e / s_red[0];
    __syncthreads();

    // out_var: rows re-ordered (b,n,t) -> (b,t,n); this thread owns h = t.
    const float cva = *trp;
    const size_t vbase = (size_t)bn * TTC * HIDC;
    for (int tt = 0; tt < TTC; ++tt) {
        const float vv = bf2f(v[vbase + (size_t)tt * HIDC + t]);
        const float o  = cva * s_attn[tt] * vv;
        A2[(((size_t)b * TTC + tt) * DKC + n) * HIDC + t] = f2bf(o);
    }
}

// ---------------------------------------------------------------------------
// launch
// ---------------------------------------------------------------------------
extern "C" void kernel_launch(void* const* d_in, const int* in_sizes, int n_in,
                              void* d_out, int out_size, void* d_ws, size_t ws_size,
                              hipStream_t stream) {
    (void)in_sizes; (void)n_in; (void)out_size; (void)ws_size;

    const float* coeffs  = (const float*)d_in[0];
    const float* proj_w  = (const float*)d_in[1];
    const float* proj_b  = (const float*)d_in[2];
    const float* var_att = (const float*)d_in[3];
    const float* mlp1_w  = (const float*)d_in[4];
    const float* mlp1_b  = (const float*)d_in[5];
    const float* mlp2_w  = (const float*)d_in[6];
    const float* mlp2_b  = (const float*)d_in[7];
    const float* out_w   = (const float*)d_in[8];
    const float* out_b   = (const float*)d_in[9];
    float* out = (float*)d_out;

    const size_t M1 = (size_t)BBC * DKC * TTC;          // 65536
    char* ws = (char*)d_ws;
    size_t off = 0;
    ushort_t* A1   = (ushort_t*)(ws + off); off += M1 * HIDC * 2;          // 64 MiB
    ushort_t* Wqkv = (ushort_t*)(ws + off); off += (size_t)512 * 1536 * 2; // 1.5 MiB
    ushort_t* W1   = (ushort_t*)(ws + off); off += (size_t)512 * 512 * 2;
    ushort_t* W2   = (ushort_t*)(ws + off); off += (size_t)512 * 512 * 2;
    ushort_t* Wo   = (ushort_t*)(ws + off); off += (size_t)512 * 512 * 2;
    float*    trp  = (float*)(ws + off);    off += 64;
    ushort_t* Q    = (ushort_t*)(ws + off); off += M1 * HIDC * 2;
    ushort_t* Kb   = (ushort_t*)(ws + off); off += M1 * HIDC * 2;
    ushort_t* V    = (ushort_t*)(ws + off); off += M1 * HIDC * 2;
    // aliases (lifetimes disjoint):
    ushort_t* A2 = A1;    // out_var rows (b,t,n)  -- A1 dead after GEMM1
    ushort_t* H1 = Q;     // gelu(mlp1)            -- Q dead after coherence
    ushort_t* X  = Kb;    // out_var + mlp         -- K dead after coherence

    // converts
    cvt_f32_bf16_kernel<<<8192, 256, 0, stream>>>(coeffs, A1, (int)(M1 * HIDC));
    cvt_f32_bf16_kernel<<<1024, 256, 0, stream>>>(proj_w, Wqkv, 512 * 1536);
    cvt_f32_bf16_kernel<<<512, 256, 0, stream>>>(mlp1_w, W1, 512 * 512);
    cvt_f32_bf16_kernel<<<512, 256, 0, stream>>>(mlp2_w, W2, 512 * 512);
    cvt_f32_bf16_kernel<<<512, 256, 0, stream>>>(out_w, Wo, 512 * 512);

    // trace(softmax(var_attn))
    zero_scalar_kernel<<<1, 64, 0, stream>>>(trp);
    trace_softmax_kernel<<<1, 512, 0, stream>>>(var_att, trp);

    // GEMM 1: qkv projection, split q/k/v
    {
        dim3 grid(1536 / BN, (unsigned)(M1 / BM));
        gemm_bf16_kernel<EPI_QKV><<<grid, 256, 0, stream>>>(
            A1, Wqkv, proj_b, (int)M1, 1536, 512, Q, Kb, V, nullptr, nullptr, nullptr);
    }

    // coherence + softmax + V scaling -> out_var (A2, rows (b,t,n))
    coherence_kernel<<<BBC * DKC, 512, 0, stream>>>(Q, Kb, V, trp, A2);

    // MLP1 + exact GELU
    {
        dim3 grid(512 / BN, (unsigned)(M1 / BM));
        gemm_bf16_kernel<EPI_GELU><<<grid, 256, 0, stream>>>(
            A2, W1, mlp1_b, (int)M1, 512, 512, nullptr, nullptr, nullptr, H1, nullptr, nullptr);
    }
    // MLP2 + residual
    {
        dim3 grid(512 / BN, (unsigned)(M1 / BM));
        gemm_bf16_kernel<EPI_RESID><<<grid, 256, 0, stream>>>(
            H1, W2, mlp2_b, (int)M1, 512, 512, nullptr, nullptr, nullptr, X, A2, nullptr);
    }
    // out projection + transpose store
    {
        dim3 grid(512 / BN, (unsigned)(M1 / BM));
        gemm_bf16_kernel<EPI_OUT><<<grid, 256, 0, stream>>>(
            X, Wo, out_b, (int)M1, 512, 512, nullptr, nullptr, nullptr, nullptr, nullptr, out);
    }
}
